// Balancer_10660108829428
// MI455X (gfx1250) — matrix operand order repaired and checked by reading, not compiled, not run on hardware
//
#include <hip/hip_runtime.h>
#include <hip/hip_bf16.h>

typedef __attribute__((ext_vector_type(16))) _Float16 v16h;
typedef __attribute__((ext_vector_type(8)))  float    v8f;

#define BATCH   64
#define IMG_H   376
#define IMG_W   1248
#define NBOX    48
#define KPAD    64          // boxes padded to 64 (2 WMMA K=32 steps)
#define W_TILES 78          // 1248 / 16
#define H_TILES 24          // ceil(376/16) -> padded to 384
#define STRIPS_PER_BATCH (W_TILES * 2)      // each strip = 12 row-tiles
#define WAVES_PER_BLOCK  8
#define BLOCKS_PER_BATCH ((STRIPS_PER_BATCH + WAVES_PER_BLOCK - 1) / WAVES_PER_BLOCK) // 20

__global__ void __launch_bounds__(256)
balancer_wmma_kernel(const float* __restrict__ loss,
                     const float* __restrict__ boxes,  // [B, N, 4] = u1,v1,u2,v2
                     float* __restrict__ acc)          // acc[0]=S_fg, acc[1]=S_tot
{
    __shared__ int u1s[KPAD], u2s[KPAD], v1s[KPAD], v2s[KPAD];

    const int batch  = blockIdx.x / BLOCKS_PER_BATCH;
    const int blkInB = blockIdx.x % BLOCKS_PER_BATCH;
    const int tid    = threadIdx.x;
    const int lane   = tid & 31;
    const int waveId = tid >> 5;

    // ---- Cache integer box bounds for this batch in LDS (pad 48..63 empty) ----
    if (tid < KPAD) {
        if (tid < NBOX) {
            const float* bp = boxes + ((size_t)batch * NBOX + tid) * 4;
            u1s[tid] = (int)floorf(bp[0]);
            v1s[tid] = (int)floorf(bp[1]);
            u2s[tid] = (int)ceilf(bp[2]);
            v2s[tid] = (int)ceilf(bp[3]);
        } else {
            u1s[tid] = 0; u2s[tid] = 0; v1s[tid] = 0; v2s[tid] = 0;  // covers nothing
        }
    }
    __syncthreads();

    float s_fg = 0.0f, s_tot = 0.0f;

    const int strip = blkInB * WAVES_PER_BLOCK + waveId;   // wave-uniform
    if (strip < STRIPS_PER_BATCH) {
        const int tile_u = strip >> 1;
        const int half   = strip & 1;                       // row-tiles [half*12, half*12+12)
        const int u0     = tile_u * 16;

        const int r     = lane & 15;           // A row index within tile
        const int koff  = (lane >> 4) * 8;     // A K-sub-offset for high half-lanes
        const int mbase = (lane >> 4) * 8;     // C row offset for high half-lanes

        // ---- Per-lane cached v-bounds for the A operand (fixed per lane) ----
        // A element e of k-step kb tests box: kb*32 + ((e<8) ? koff+e : 16+koff+e-8)
        int vb0[16], vb1[16];
        #pragma unroll
        for (int e = 0; e < 16; ++e) {
            int kl  = (e < 8) ? (koff + e) : (16 + koff + (e - 8));
            vb0[e] = (v1s[kl]      & 0xffff) | (v2s[kl]      << 16);
            vb1[e] = (v1s[32 + kl] & 0xffff) | (v2s[32 + kl] << 16);
        }

        // ---- Build B once per strip: lane%16 <-> N (column), elements <-> K (same order as A) ----
        const int u = u0 + (lane & 15);
        v16h b0, b1;
        #pragma unroll
        for (int e = 0; e < 16; ++e) {
            int kl = (e < 8) ? (koff + e) : (16 + koff + (e - 8));
            b0[e] = (u >= u1s[kl] && u < u2s[kl]) ? (_Float16)1.0f : (_Float16)0.0f;
            b1[e] = (u >= u1s[32 + kl] && u < u2s[32 + kl]) ? (_Float16)1.0f : (_Float16)0.0f;
        }

        const int n = lane & 15;               // C column within tile
        // Per-wave base pointer for this column of loss values.
        const float* colp = loss + (size_t)batch * IMG_H * IMG_W + (u0 + n);

        for (int t = 0; t < 12; ++t) {
            const int v0   = (half * 12 + t) * 16;
            const int vrow = v0 + r;

            // ---- Build A for both K halves (register-cached bounds, compares only) ----
            v16h a0, a1;
            #pragma unroll
            for (int e = 0; e < 16; ++e) {
                int w0 = vb0[e], w1 = vb1[e];
                a0[e] = (vrow >= (w0 & 0xffff) && vrow < (w0 >> 16)) ? (_Float16)1.0f : (_Float16)0.0f;
                a1[e] = (vrow >= (w1 & 0xffff) && vrow < (w1 >> 16)) ? (_Float16)1.0f : (_Float16)0.0f;
            }

            // ---- Rank-64 rasterization: counts tile = A x B (two K=32 steps) ----
            v8f c = {};
            c = __builtin_amdgcn_wmma_f32_16x16x32_f16(false, a0, false, b0,
                                                       (short)0, c, false, false);
            c = __builtin_amdgcn_wmma_f32_16x16x32_f16(false, a1, false, b1,
                                                       (short)0, c, false, false);

            // ---- Branchless masked accumulation ----
            // Rows touched by this lane are vgbase..vgbase+7 (vgbase multiple of 8):
            // either all in-bounds (vgbase <= 368) or all out (vgbase == 376).
            // Clamp the BASE so all 8 loads stay in-bounds with constant byte
            // offsets (j*4992 fits the signed-24-bit IOFFSET), and apply one
            // validity select to the tile-partial sums.
            const int vgbase = v0 + mbase;
            const int vgb    = vgbase < (IMG_H - 8) ? vgbase : (IMG_H - 8);
            const float* p   = colp + (size_t)vgb * IMG_W;

            float ttot = 0.0f, tfg = 0.0f;
            #pragma unroll
            for (int j = 0; j < 8; ++j) {
                float lv = p[j * IMG_W];                 // immediate-offset loads
                ttot += lv;
                tfg  += (c[j] > 0.0f) ? lv : 0.0f;       // cndmask, not branch
            }
            const bool vok = vgbase < IMG_H;             // uniform across j
            s_tot += vok ? ttot : 0.0f;
            s_fg  += vok ? tfg  : 0.0f;
        }
    }

    // ---- Wave32 reduction, then two global f32 atomics per wave ----
    #pragma unroll
    for (int off = 16; off >= 1; off >>= 1) {
        s_fg  += __shfl_xor(s_fg,  off, 32);
        s_tot += __shfl_xor(s_tot, off, 32);
    }
    if (lane == 0) {
        atomicAdd(acc + 0, s_fg);
        atomicAdd(acc + 1, s_tot);
    }
}

__global__ void balancer_zero_kernel(float* acc) {
    if (threadIdx.x < 2) acc[threadIdx.x] = 0.0f;
}

__global__ void balancer_finalize_kernel(const float* __restrict__ acc,
                                         float* __restrict__ out) {
    const float P = (float)((size_t)BATCH * IMG_H * IMG_W);
    float s_fg  = acc[0];
    float s_tot = acc[1];
    float fg_loss = 13.0f * s_fg / P;
    float bg_loss = (s_tot - s_fg) / P;
    out[0] = fg_loss + bg_loss;
    out[1] = fg_loss;
    out[2] = bg_loss;
}

extern "C" void kernel_launch(void* const* d_in, const int* in_sizes, int n_in,
                              void* d_out, int out_size, void* d_ws, size_t ws_size,
                              hipStream_t stream) {
    const float* loss  = (const float*)d_in[0];
    const float* boxes = (const float*)d_in[1];
    float* out = (float*)d_out;
    float* acc = (float*)d_ws;

    balancer_zero_kernel<<<1, 32, 0, stream>>>(acc);

    dim3 grid(BATCH * BLOCKS_PER_BATCH);   // 64 * 20 = 1280 blocks
    dim3 block(256);                       // 8 waves/block (wave32)
    balancer_wmma_kernel<<<grid, block, 0, stream>>>(loss, boxes, acc);

    balancer_finalize_kernel<<<1, 1, 0, stream>>>(acc, out);
}
